// GDSRec_61323543052500
// MI455X (gfx1250) — matrix operand order repaired and checked by reading, not compiled, not run on hardware
//
#include <hip/hip_runtime.h>

// ---------------------------------------------------------------------------
// GDSRec on gfx1250: f16 WMMA (16x16x32) GEMMs out of LDS, f32 accumulate.
// Transcendentals via CDNA5 hardware TRANS ops (v_tanh_f32 / v_exp_f32).
// A-fragment loads hoisted ahead of the WMMA chain per stage so ds_load_b128s
// pipeline instead of load->wait->wmma serialization.
// ---------------------------------------------------------------------------

typedef _Float16 h8   __attribute__((ext_vector_type(8)));
typedef _Float16 v16h __attribute__((ext_vector_type(16)));
typedef float    v8f  __attribute__((ext_vector_type(8)));

union F16x16 { v16h v; h8 h[2]; };

#define LDA 136   // 128-col A buffer row stride (halves), padded for LDS banks
#define LDX 72    // 64-col  x buffer row stride (halves)
#define EPSV 1e-10f

// Hardware tanh (CDNA5 TRANS32 op).
__device__ __forceinline__ float fast_tanh(float x) {
#if __has_builtin(__builtin_amdgcn_tanhf)
    return __builtin_amdgcn_tanhf(x);
#elif __has_builtin(__builtin_amdgcn_tanh_f32)
    return __builtin_amdgcn_tanh_f32(x);
#else
    float r;
    asm volatile("v_tanh_f32 %0, %1\n\tv_nop" : "=v"(r) : "v"(x));
    return r;
#endif
}

// Branch-free exp via v_exp_f32 (exp2) with log2(e) prescale.
__device__ __forceinline__ float fast_exp(float x) {
    return __expf(x);
}

// Build a 16x32 f16 A-fragment from an LDS row-major buffer.
// Layout (ISA 7.12.2): lanes 0-15: M=lane, K = c*32 + {0..7, 16..23}
//                      lanes 16-31: M=lane-16, K = c*32 + {8..15, 24..31}
__device__ __forceinline__ v16h ld_afrag(const _Float16* buf, int ld, int lane, int chunk) {
    const int row = lane & 15;
    const int kb  = chunk * 32 + ((lane < 16) ? 0 : 8);
    const _Float16* p = buf + row * ld + kb;
    F16x16 a;
    a.h[0] = *reinterpret_cast<const h8*>(p);
    a.h[1] = *reinterpret_cast<const h8*>(p + 16);
    return a.v;
}

// Pre-packed weight fragment: [chunk][ntile][lane][16 halves], contiguous.
__device__ __forceinline__ v16h ld_wfrag(const _Float16* base, int chunk, int wid, int lane) {
    return *reinterpret_cast<const v16h*>(base + (((chunk * 4 + wid) * 32 + lane) << 4));
}

// ---------------------------------------------------------------------------
// Pack a f32 [K x 64] weight matrix into the B-fragment register image.
//   K = c*32 + (lane<16 ? 0 : 8) + (i<8 ? i : i+8) ; N = t*16 + (lane&15)
// ---------------------------------------------------------------------------
__global__ __launch_bounds__(256) void gdsrec_pack_kernel(
    const float* __restrict__ W, _Float16* __restrict__ dst, int nchunks) {
    int total = nchunks * 4 * 32 * 16;
    for (int o = blockIdx.x * blockDim.x + threadIdx.x; o < total;
         o += gridDim.x * blockDim.x) {
        int i = o & 15;
        int l = (o >> 4) & 31;
        int t = (o >> 9) & 3;
        int c = o >> 11;
        int K = c * 32 + ((l < 16) ? 0 : 8) + ((i < 8) ? i : i + 8);
        int n = t * 16 + (l & 15);
        dst[o] = (_Float16)W[K * 64 + n];
    }
}

// ---------------------------------------------------------------------------
// Neighbor aggregation. One workgroup (4 waves) per segment; rows tiled by 16.
// frag layout per block: g_w1 @ 0 (4 chunks), g_w2 @ 8192 (2), a_w1_top @ 12288 (2)
// ---------------------------------------------------------------------------
__global__ __launch_bounds__(128) void gdsrec_agg_kernel(
    const int*   __restrict__ pair, int N, int ntiles,
    const float* __restrict__ neigh_tab, const float* __restrict__ rate_tab,
    const float* __restrict__ center_tab, const int* __restrict__ center_idx, int cstride,
    const _Float16* __restrict__ frag,
    const float* __restrict__ g_b1, const float* __restrict__ g_b2,
    const float* __restrict__ a_w1, const float* __restrict__ a_b1,
    const float* __restrict__ a_w2, const float* __restrict__ a_b2,
    const float* __restrict__ agg_w, const float* __restrict__ agg_b,
    float* __restrict__ out) {

    __shared__ _Float16 sA [16 * LDA];
    __shared__ _Float16 sX1[16 * LDX];
    __shared__ _Float16 sX2[16 * LDX];
    __shared__ float sCW[64];     // center @ a_w1[64:128]
    __shared__ float sS[64];      // un-normalized weighted sum
    __shared__ float sMask[16];
    __shared__ int   sIdx[32];
    __shared__ float sAacc[16];
    __shared__ float sAvals[16];
    __shared__ float sTot;

    const int tid  = threadIdx.x;
    const int lane = tid & 31;
    const int wid  = tid >> 5;          // ntile owned by this wave
    const int seg  = blockIdx.x;

    // Resident per-wave weight fragments for column tile `wid`.
    v16h g1f[4], g2f[2], a1f[2];
#pragma unroll
    for (int c = 0; c < 4; ++c) g1f[c] = ld_wfrag(frag,         c, wid, lane);
#pragma unroll
    for (int c = 0; c < 2; ++c) g2f[c] = ld_wfrag(frag + 8192,  c, wid, lane);
#pragma unroll
    for (int c = 0; c < 2; ++c) a1f[c] = ld_wfrag(frag + 12288, c, wid, lane);

    // Per-segment precompute: cw[n] = sum_k center[k] * a_w1[(64+k)*64 + n]
    if (tid < 64) {
        const int cidx = center_idx[(size_t)seg * cstride];
        const float* cen = center_tab + (size_t)cidx * 64;
        float acc = 0.f;
#pragma unroll 8
        for (int k = 0; k < 64; ++k) acc += cen[k] * a_w1[(64 + k) * 64 + tid];
        sCW[tid] = acc;
        sS[tid]  = 0.f;
    }
    if (tid == 0) sTot = 0.f;
    __syncthreads();

    const int ncol  = wid * 16 + (lane & 15);
    const int mbase = (lane < 16) ? 0 : 8;
    const float b1v = g_b1[ncol], b2v = g_b2[ncol];
    const float ab1 = a_b1[ncol];
    const float w2v = a_w2[ncol];
    const float ab2 = a_b2[0];

    for (int tile = 0; tile < ntiles; ++tile) {
        const int row0 = tile * 16;
        if (tid < 16) {
            int r = row0 + tid, i0 = 0, i1 = 0;
            if (r < N) {
                i0 = pair[((size_t)seg * N + r) * 2];
                i1 = pair[((size_t)seg * N + r) * 2 + 1];
            }
            sIdx[tid * 2] = i0; sIdx[tid * 2 + 1] = i1;
            sMask[tid] = (i0 > 0) ? 1.f : 0.f;
            sAacc[tid] = 0.f;
        }
        __syncthreads();

        // Gather concat(neigh_emb, rate_emb) -> sA (f16), 16 cols per thread.
        {
            const int row = tid >> 3, c0 = (tid & 7) * 16;
            const int i0 = sIdx[row * 2], i1 = sIdx[row * 2 + 1];
            const float* src = (c0 < 64) ? (neigh_tab + (size_t)i0 * 64 + c0)
                                         : (rate_tab  + (size_t)i1 * 64 + (c0 - 64));
            _Float16* dst = &sA[row * LDA + c0];
#pragma unroll
            for (int j = 0; j < 16; ++j) dst[j] = (_Float16)src[j];
        }
        __syncthreads();

        // g layer 1: [16x128]@[128x64], tanh. Load all A fragments first.
        {
            v16h a0 = ld_afrag(sA, LDA, lane, 0);
            v16h a1 = ld_afrag(sA, LDA, lane, 1);
            v16h a2 = ld_afrag(sA, LDA, lane, 2);
            v16h a3 = ld_afrag(sA, LDA, lane, 3);
            v8f acc = {0.f, 0.f, 0.f, 0.f, 0.f, 0.f, 0.f, 0.f};
            acc = __builtin_amdgcn_wmma_f32_16x16x32_f16(false, a0, false, g1f[0],
                                                         (short)0, acc, false, false);
            acc = __builtin_amdgcn_wmma_f32_16x16x32_f16(false, a1, false, g1f[1],
                                                         (short)0, acc, false, false);
            acc = __builtin_amdgcn_wmma_f32_16x16x32_f16(false, a2, false, g1f[2],
                                                         (short)0, acc, false, false);
            acc = __builtin_amdgcn_wmma_f32_16x16x32_f16(false, a3, false, g1f[3],
                                                         (short)0, acc, false, false);
#pragma unroll
            for (int v = 0; v < 8; ++v)
                sX1[(v + mbase) * LDX + ncol] = (_Float16)fast_tanh(acc[v] + b1v);
        }
        __syncthreads();

        // g layer 2: x = h1 @ g_w2 + b2 (no activation)
        {
            v16h a0 = ld_afrag(sX1, LDX, lane, 0);
            v16h a1 = ld_afrag(sX1, LDX, lane, 1);
            v8f xacc = {0.f, 0.f, 0.f, 0.f, 0.f, 0.f, 0.f, 0.f};
            xacc = __builtin_amdgcn_wmma_f32_16x16x32_f16(false, a0, false, g2f[0],
                                                          (short)0, xacc, false, false);
            xacc = __builtin_amdgcn_wmma_f32_16x16x32_f16(false, a1, false, g2f[1],
                                                          (short)0, xacc, false, false);
#pragma unroll
            for (int v = 0; v < 8; ++v)
                sX2[(v + mbase) * LDX + ncol] = (_Float16)(xacc[v] + b2v);
        }
        __syncthreads();

        // attention layer 1: x @ a_w1_top + mask*cw + b1, relu; dot with a_w2
        {
            v16h a0 = ld_afrag(sX2, LDX, lane, 0);
            v16h a1 = ld_afrag(sX2, LDX, lane, 1);
            v8f aacc = {0.f, 0.f, 0.f, 0.f, 0.f, 0.f, 0.f, 0.f};
            aacc = __builtin_amdgcn_wmma_f32_16x16x32_f16(false, a0, false, a1f[0],
                                                          (short)0, aacc, false, false);
            aacc = __builtin_amdgcn_wmma_f32_16x16x32_f16(false, a1, false, a1f[1],
                                                          (short)0, aacc, false, false);
            const float cwv = sCW[ncol];
#pragma unroll
            for (int v = 0; v < 8; ++v) {
                const int m = v + mbase;
                float h = aacc[v] + sMask[m] * cwv + ab1;
                h = fmaxf(h, 0.f);
                atomicAdd(&sAacc[m], h * w2v);
            }
        }
        __syncthreads();

        if (tid < 16) {
            float a = fast_exp(sAacc[tid] + ab2) * sMask[tid];
            sAvals[tid] = a;
            atomicAdd(&sTot, a);
        }
        __syncthreads();

        // un-normalized weighted sum: s[n] += sum_r a_r * x[r][n]
        if (tid < 64) {
            float sum = 0.f;
#pragma unroll
            for (int r = 0; r < 16; ++r)
                sum += sAvals[r] * (float)sX2[r * LDX + tid];
            sS[tid] += sum;
        }
        __syncthreads();
    }

    // out = tanh((s / (tot+eps)) @ agg_w + agg_b)
    if (tid < 64) {
        const float inv = 1.f / (sTot + EPSV);
        float acc = agg_b[tid];
#pragma unroll 8
        for (int k = 0; k < 64; ++k) acc += (sS[k] * inv) * agg_w[k * 64 + tid];
        out[(size_t)seg * 64 + tid] = fast_tanh(acc);
    }
}

// ---------------------------------------------------------------------------
// Prediction MLP over 13,056 rows (256 of concat(h_i,z_j) + 12800 of
// concat(y_i, z_j[b])). 16 rows per workgroup.
// ---------------------------------------------------------------------------
__global__ __launch_bounds__(128) void gdsrec_pred_kernel(
    const float* __restrict__ h_i, const float* __restrict__ z_j,
    const float* __restrict__ y_i,
    const _Float16* __restrict__ fw1, const _Float16* __restrict__ fw2,
    const float* __restrict__ b1, const float* __restrict__ b2,
    const float* __restrict__ w3, const float* __restrict__ b3,
    float* __restrict__ p) {

    __shared__ _Float16 sA [16 * LDA];
    __shared__ _Float16 sX1[16 * LDX];
    __shared__ float sAcc[16];

    const int tid  = threadIdx.x;
    const int lane = tid & 31;
    const int wid  = tid >> 5;

    v16h w1f[4], w2f[2];
#pragma unroll
    for (int c = 0; c < 4; ++c) w1f[c] = ld_wfrag(fw1, c, wid, lane);
#pragma unroll
    for (int c = 0; c < 2; ++c) w2f[c] = ld_wfrag(fw2, c, wid, lane);

    if (tid < 16) sAcc[tid] = 0.f;

    // Gather concat(A_row, z_row) -> sA
    {
        const int row = tid >> 3, c0 = (tid & 7) * 16;
        const int r = blockIdx.x * 16 + row;
        const float* src0;
        int bidx;
        if (r < 256) { src0 = h_i + (size_t)r * 64; bidx = r; }
        else { const int rr = r - 256; src0 = y_i + (size_t)rr * 64; bidx = rr / 50; }
        const float* src = (c0 < 64) ? (src0 + c0)
                                     : (z_j + (size_t)bidx * 64 + (c0 - 64));
        _Float16* dst = &sA[row * LDA + c0];
#pragma unroll
        for (int j = 0; j < 16; ++j) dst[j] = (_Float16)src[j];
    }
    __syncthreads();

    const int ncol  = wid * 16 + (lane & 15);
    const int mbase = (lane < 16) ? 0 : 8;

    {
        v16h a0 = ld_afrag(sA, LDA, lane, 0);
        v16h a1 = ld_afrag(sA, LDA, lane, 1);
        v16h a2 = ld_afrag(sA, LDA, lane, 2);
        v16h a3 = ld_afrag(sA, LDA, lane, 3);
        v8f acc = {0.f, 0.f, 0.f, 0.f, 0.f, 0.f, 0.f, 0.f};
        acc = __builtin_amdgcn_wmma_f32_16x16x32_f16(false, a0, false, w1f[0],
                                                     (short)0, acc, false, false);
        acc = __builtin_amdgcn_wmma_f32_16x16x32_f16(false, a1, false, w1f[1],
                                                     (short)0, acc, false, false);
        acc = __builtin_amdgcn_wmma_f32_16x16x32_f16(false, a2, false, w1f[2],
                                                     (short)0, acc, false, false);
        acc = __builtin_amdgcn_wmma_f32_16x16x32_f16(false, a3, false, w1f[3],
                                                     (short)0, acc, false, false);
        const float bv = b1[ncol];
#pragma unroll
        for (int v = 0; v < 8; ++v)
            sX1[(v + mbase) * LDX + ncol] = (_Float16)fast_tanh(acc[v] + bv);
    }
    __syncthreads();

    {
        v16h a0 = ld_afrag(sX1, LDX, lane, 0);
        v16h a1 = ld_afrag(sX1, LDX, lane, 1);
        v8f acc2 = {0.f, 0.f, 0.f, 0.f, 0.f, 0.f, 0.f, 0.f};
        acc2 = __builtin_amdgcn_wmma_f32_16x16x32_f16(false, a0, false, w2f[0],
                                                      (short)0, acc2, false, false);
        acc2 = __builtin_amdgcn_wmma_f32_16x16x32_f16(false, a1, false, w2f[1],
                                                      (short)0, acc2, false, false);
        const float bv = b2[ncol], wv = w3[ncol];
#pragma unroll
        for (int v = 0; v < 8; ++v) {
            float h = fast_tanh(acc2[v] + bv);
            atomicAdd(&sAcc[v + mbase], h * wv);
        }
    }
    __syncthreads();

    if (tid < 16) p[blockIdx.x * 16 + tid] = sAcc[tid] + b3[0];
}

// ---------------------------------------------------------------------------
// Final: r1, masked mean of social preds, combine with averages.
// ---------------------------------------------------------------------------
__global__ __launch_bounds__(256) void gdsrec_final_kernel(
    const float* __restrict__ p, const int* __restrict__ u_user_pad,
    const float* __restrict__ u_avgs, const float* __restrict__ i_avgs,
    float* __restrict__ out) {
    const int b = threadIdx.x;
    const float r1 = p[b];
    float sum = 0.f, msum = 0.f;
    for (int u = 0; u < 50; ++u) {
        const float m = (float)u_user_pad[(b * 50 + u) * 2 + 1];
        sum  += p[256 + b * 50 + u] * m;
        msum += m;
    }
    const float r2 = sum / (msum + EPSV);
    out[b] = 0.5f * (r1 + r2) + 0.5f * (u_avgs[b] + i_avgs[b]);
}

// ---------------------------------------------------------------------------
extern "C" void kernel_launch(void* const* d_in, const int* in_sizes, int n_in,
                              void* d_out, int out_size, void* d_ws, size_t ws_size,
                              hipStream_t stream) {
    (void)in_sizes; (void)n_in; (void)out_size; (void)ws_size;

    const int*   uids        = (const int*)  d_in[0];
    const int*   iids        = (const int*)  d_in[1];
    const int*   u_itemdiv   = (const int*)  d_in[2];
    const float* u_avgs      = (const float*)d_in[3];
    const int*   u_user      = (const int*)  d_in[4];
    const int*   u_user_item = (const int*)  d_in[5];
    const int*   i_user      = (const int*)  d_in[6];
    const float* i_avgs      = (const float*)d_in[7];
    const float* user_emb    = (const float*)d_in[8];
    const float* item_emb    = (const float*)d_in[9];
    const float* rate_emb    = (const float*)d_in[10];
    // blocks: user @11, item @21, social @31; within a block:
    //   +0 g_w1, +1 g_b1, +2 g_w2, +3 g_b2, +4 a_w1, +5 a_b1, +6 a_w2, +7 a_b2,
    //   +8 agg_w, +9 agg_b.  pred: w1@41 b1@42 w2@43 b2@44 w3@45 b3@46.
    auto F = [&](int i) { return (const float*)d_in[i]; };

    // Workspace: packed f16 fragments, then f32 intermediates.
    _Float16* frag = (_Float16*)d_ws;
    const size_t BLOCK = 16384; // halves per agg block (g1 8192 + g2 4096 + a1 4096)
    _Float16* bf[3] = { frag, frag + BLOCK, frag + 2 * BLOCK };
    _Float16* pw1 = frag + 3 * BLOCK;
    _Float16* pw2 = pw1 + 8192;
    float* fbase = (float*)(frag + 3 * BLOCK + 12288); // byte off 122880, 16B aligned
    float* h_i = fbase;
    float* z_j = h_i + 256 * 64;
    float* y_i = z_j + 256 * 64;
    float* p   = y_i + 12800 * 64;

    const int bases[3] = {11, 21, 31};
    for (int i = 0; i < 3; ++i) {
        gdsrec_pack_kernel<<<8, 256, 0, stream>>>(F(bases[i] + 0), bf[i],          4); // g_w1
        gdsrec_pack_kernel<<<4, 256, 0, stream>>>(F(bases[i] + 2), bf[i] + 8192,   2); // g_w2
        gdsrec_pack_kernel<<<4, 256, 0, stream>>>(F(bases[i] + 4), bf[i] + 12288,  2); // a_w1 top
    }
    gdsrec_pack_kernel<<<8, 256, 0, stream>>>(F(41), pw1, 4);
    gdsrec_pack_kernel<<<4, 256, 0, stream>>>(F(43), pw2, 2);

    // user model: items rated by each user -> h_i (256x64)
    gdsrec_agg_kernel<<<256, 128, 0, stream>>>(
        u_itemdiv, 200, 13, item_emb, rate_emb, user_emb, uids, 1, bf[0],
        F(12), F(14), F(15), F(16), F(17), F(18), F(19), F(20), h_i);
    // item model: users who rated each item -> z_j (256x64)
    gdsrec_agg_kernel<<<256, 128, 0, stream>>>(
        i_user, 200, 13, user_emb, rate_emb, item_emb, iids, 1, bf[1],
        F(22), F(24), F(25), F(26), F(27), F(28), F(29), F(30), z_j);
    // social model: per (b,u) neighbor, their items -> y_i (12800x64)
    gdsrec_agg_kernel<<<12800, 128, 0, stream>>>(
        u_user_item, 30, 2, item_emb, rate_emb, user_emb, u_user, 2, bf[2],
        F(32), F(34), F(35), F(36), F(37), F(38), F(39), F(40), y_i);

    // prediction MLP over 256 + 12800 = 13056 rows (816 tiles of 16)
    gdsrec_pred_kernel<<<816, 128, 0, stream>>>(
        h_i, z_j, y_i, pw1, pw2, F(42), F(44), F(45), F(46), p);

    gdsrec_final_kernel<<<1, 256, 0, stream>>>(p, u_user, u_avgs, i_avgs,
                                               (float*)d_out);
}
